// selfAttentionProcessorInit_55379308315027
// MI455X (gfx1250) — matrix operand order repaired and checked by reading, not compiled, not run on hardware
//
#include <hip/hip_runtime.h>
#include <math.h>

// MI455X (gfx1250): wave32, WMMA 16x16x32 f16 -> f32 accum.
// B=2, S=2048, C=640, H=8, D=80 (padded to 96 for the QK^T K-dim).

typedef __attribute__((ext_vector_type(16))) _Float16 v16h;
typedef __attribute__((ext_vector_type(8)))  float    v8f;

#define WMMA_F32_F16(a, b, c) \
  __builtin_amdgcn_wmma_f32_16x16x32_f16(false, (a), false, (b), (short)0, (c), false, false)

// IGroupLP scheduling directives: 0x008 = MFMA/WMMA group, 0x020 = VMEM read group.
#if __has_builtin(__builtin_amdgcn_sched_group_barrier)
#define SCHED_VMEM(n) __builtin_amdgcn_sched_group_barrier(0x020, (n), 0)
#define SCHED_WMMA(n) __builtin_amdgcn_sched_group_barrier(0x008, (n), 0)
#else
#define SCHED_VMEM(n)
#define SCHED_WMMA(n)
#endif

// ---- fragment loaders (contiguous k order: lane half h holds k 16h..16h+15, wave32) ----

// A (16x32, MxK): lane<16 -> row (lane), K k0..k0+15;
//                 lane>=16 -> row (lane-16), K k0+16..k0+31.  One 32B load.
static __device__ __forceinline__ v16h load_a_f16(const _Float16* __restrict__ X, int ld, int m0, int k0) {
  const int l = threadIdx.x & 31;
  const _Float16* p = X + (m0 + (l & 15)) * ld + k0 + ((l >> 4) * 16);
  v16h a;
#pragma unroll
  for (int i = 0; i < 16; ++i) a[i] = p[i];
  return a;
}

// B (32x16, KxN): lane<16 -> col N=(lane), K=k0..k0+15; lane>=16 -> col (lane-16), K=k0+16..31.
// B columns stored as contiguous f16 rows of T: B[k][n] = T[n_row0 + n][k0 + k].  One 32B load.
static __device__ __forceinline__ v16h load_b_f16_rows(const _Float16* __restrict__ T, int ld, int n_row0, int k0) {
  const int l = threadIdx.x & 31;
  const _Float16* p = T + (n_row0 + (l & 15)) * ld + k0 + ((l >> 4) * 16);
  v16h b;
#pragma unroll
  for (int i = 0; i < 16; ++i) b[i] = p[i];
  return b;
}

// ---- prep 1: X f32 -> f16, vectorized 4/thread ----
__global__ __launch_bounds__(256) void cvt_x_kernel(const float* __restrict__ X,
                                                    _Float16* __restrict__ x16) {
  const int i = (blockIdx.x * 256 + threadIdx.x) * 4;
  const float4 v = *(const float4*)(X + i);
  _Float16* o = x16 + i;
  o[0] = (_Float16)v.x; o[1] = (_Float16)v.y; o[2] = (_Float16)v.z; o[3] = (_Float16)v.w;
}

// ---- prep 2: transpose+convert weights: W[k][n] f32 -> Wt[n][k] f16 (LDS-tiled) ----
__global__ __launch_bounds__(256) void wt_kernel(const float* __restrict__ Wq,
    const float* __restrict__ Wk, const float* __restrict__ Wv, const float* __restrict__ Wo,
    _Float16* __restrict__ wt) {
  __shared__ float tile[16][17];
  const float* W = (blockIdx.z == 0) ? Wq : (blockIdx.z == 1) ? Wk : (blockIdx.z == 2) ? Wv : Wo;
  _Float16* out = wt + blockIdx.z * 640 * 640;
  const int n0 = blockIdx.x * 16, k0 = blockIdx.y * 16;
  tile[threadIdx.y][threadIdx.x] = W[(k0 + threadIdx.y) * 640 + n0 + threadIdx.x];
  __syncthreads();
  out[(n0 + threadIdx.y) * 640 + k0 + threadIdx.x] = (_Float16)tile[threadIdx.x][threadIdx.y];
}

// ---- fully-unrolled pipelined 16x64-per-wave GEMM core ----
// 20 k-steps of 32, straight-line (SSA: no phi copies, no WMMA hazard nops).
// Per step: 10 b128 loads for step k+1 + 4 WMMAs for step k; sched_group_barrier
// pins the interleave so loads stay a full step ahead of their consumers.
static __device__ __forceinline__ void gemm_16x64(const _Float16* __restrict__ A, int m0,
                                                  const _Float16* __restrict__ Bt, int n0,
                                                  v8f acc[4]) {
  v16h a_cur = load_a_f16(A, 640, m0, 0);
  v16h b_cur[4];
#pragma unroll
  for (int j = 0; j < 4; ++j) b_cur[j] = load_b_f16_rows(Bt, 640, n0 + j * 16, 0);
#pragma unroll
  for (int kk = 0; kk < 20; ++kk) {
    v16h a_nxt, b_nxt[4];
    if (kk < 19) {
      a_nxt = load_a_f16(A, 640, m0, (kk + 1) * 32);
#pragma unroll
      for (int j = 0; j < 4; ++j) b_nxt[j] = load_b_f16_rows(Bt, 640, n0 + j * 16, (kk + 1) * 32);
    }
#pragma unroll
    for (int j = 0; j < 4; ++j) acc[j] = WMMA_F32_F16(a_cur, b_cur[j], acc[j]);
    if (kk < 19) {
      SCHED_VMEM(3); SCHED_WMMA(1);
      SCHED_VMEM(3); SCHED_WMMA(1);
      SCHED_VMEM(2); SCHED_WMMA(1);
      SCHED_VMEM(2); SCHED_WMMA(1);
    }
    a_cur = a_nxt;
#pragma unroll
    for (int j = 0; j < 4; ++j) b_cur[j] = b_nxt[j];
  }
}

// ---- kernel 1: fused QKV projection ----
// Q,K: [B,H,S,96] row-major (d 80..95 pre-zeroed).  V: transposed [B,H,96,S].
__global__ __launch_bounds__(128) void qkv_kernel(const _Float16* __restrict__ x16,
    const _Float16* __restrict__ wt,
    _Float16* __restrict__ q16, _Float16* __restrict__ k16, _Float16* __restrict__ vt16) {
  const int m0 = blockIdx.x * 64 + threadIdx.y * 16;   // token tile (4096 rows)
  const int g = blockIdx.y;                            // 0..29: {Q,K,V} x 10 groups of 64 cols
  const int mat = g / 10;
  const int n0 = (g % 10) * 64;

  v8f acc[4] = {};
  gemm_16x64(x16, m0, wt + mat * 640 * 640, n0, acc);

  // C/D layout: VGPR r, lane half -> (row = m0 + half*8 + r, col = lane%16)
  const int l = threadIdx.x & 31;
#pragma unroll
  for (int j = 0; j < 4; ++j) {
#pragma unroll
    for (int r = 0; r < 8; ++r) {
      const int tok = m0 + ((l >> 4) * 8) + r;
      const int c = n0 + j * 16 + (l & 15);
      const int bb = tok >> 11, s = tok & 2047;
      const int h = c / 80, d = c % 80;
      const _Float16 v = (_Float16)acc[j][r];
      if (mat == 0)      q16[((bb * 8 + h) * 2048 + s) * 96 + d] = v;
      else if (mat == 1) k16[((bb * 8 + h) * 2048 + s) * 96 + d] = v;
      else               vt16[((bb * 8 + h) * 96 + d) * 2048 + s] = v;
    }
  }
}

// ---- kernel 2: flash attention, one wave per 16-query tile, probs never hit HBM ----
__global__ __launch_bounds__(128) void attn_kernel(const _Float16* __restrict__ q16,
    const _Float16* __restrict__ k16, const _Float16* __restrict__ vt16,
    _Float16* __restrict__ ctx) {
  __shared__ _Float16 ldsP[4][16 * 32];                // per-wave P-tile transpose buffer
  const int wave = blockIdx.x * 4 + threadIdx.y;       // 0..2047 = B*H*(S/16)
  const int bh = wave >> 7;
  const int q0 = (wave & 127) * 16;
  const int b = bh >> 3, h = bh & 7;
  const _Float16* Q  = q16 + (bh * 2048 + q0) * 96;
  const _Float16* K  = k16 + bh * 2048 * 96;
  const _Float16* Vt = vt16 + bh * 96 * 2048;
  const int l = threadIdx.x & 31;
  const float scale = 0.11180339887498949f;            // 80^-0.5

  float mrow[8], lrow[8];
  v8f acc[5] = {};
#pragma unroll
  for (int r = 0; r < 8; ++r) { mrow[r] = -1e30f; lrow[r] = 0.f; }

  _Float16* P = ldsP[threadIdx.y];

  v16h aq[3];                                          // Q fragments, K-dim 96 = 3 chunks
#pragma unroll
  for (int kc = 0; kc < 3; ++kc) aq[kc] = load_a_f16(Q, 96, 0, kc * 32);

  // K fragment pipeline: kb[kc][0/1] = keys kt..kt+15 / kt+16..kt+31, dims kc*32..+31
  v16h kb[3][2];
#pragma unroll
  for (int kc = 0; kc < 3; ++kc) {
    kb[kc][0] = load_b_f16_rows(K, 96, 0,  kc * 32);
    kb[kc][1] = load_b_f16_rows(K, 96, 16, kc * 32);
  }

  for (int kt = 0; kt < 2048; kt += 32) {              // 32 keys per step
    if (kt + 64 < 2048) {                              // deep prefetch (global_prefetch_b8)
      __builtin_prefetch(K + (kt + 64 + l) * 96, 0, 3);
      __builtin_prefetch(Vt + (l * 3) * 2048 + kt + 64, 0, 3);
    }
    // V fragments for this step: issued early, consumed after softmax
    v16h bv[5];
#pragma unroll
    for (int tt = 0; tt < 5; ++tt) bv[tt] = load_b_f16_rows(Vt, 2048, tt * 16, kt);

    v8f s0 = {}, s1 = {};
#pragma unroll
    for (int kc = 0; kc < 3; ++kc) {
      s0 = WMMA_F32_F16(aq[kc], kb[kc][0], s0);
      s1 = WMMA_F32_F16(aq[kc], kb[kc][1], s1);
    }
    // refill K pipeline for kt+32 (loads overwrite consumed regs; no copies)
    if (kt + 32 < 2048) {
#pragma unroll
      for (int kc = 0; kc < 3; ++kc) {
        kb[kc][0] = load_b_f16_rows(K, 96, kt + 32, kc * 32);
        kb[kc][1] = load_b_f16_rows(K, 96, kt + 48, kc * 32);
      }
    }
    // online softmax; C-layout rows live in 16-lane halves -> xor-shuffle reductions
#pragma unroll
    for (int r = 0; r < 8; ++r) {
      float t0 = s0[r] * scale, t1 = s1[r] * scale;
      float tm = fmaxf(t0, t1);
#pragma unroll
      for (int off = 1; off < 16; off <<= 1) tm = fmaxf(tm, __shfl_xor(tm, off, 32));
      const float mn = fmaxf(mrow[r], tm);
      const float f  = __expf(mrow[r] - mn);
      const float p0 = __expf(t0 - mn), p1 = __expf(t1 - mn);
      float ps = p0 + p1;
#pragma unroll
      for (int off = 1; off < 16; off <<= 1) ps += __shfl_xor(ps, off, 32);
      lrow[r] = lrow[r] * f + ps;
      mrow[r] = mn;
#pragma unroll
      for (int tt = 0; tt < 5; ++tt) acc[tt][r] = acc[tt][r] * f;
      const int row = ((l >> 4) * 8 + r) * 32;         // C-layout -> LDS row-major P[16][32]
      P[row + (l & 15)]      = (_Float16)p0;
      P[row + 16 + (l & 15)] = (_Float16)p1;
    }
    // O += P(16x32) x V(32x80); per-wave DS ordering guarantees store->load visibility
    v16h ap = load_a_f16(P, 32, 0, 0);
#pragma unroll
    for (int tt = 0; tt < 5; ++tt) acc[tt] = WMMA_F32_F16(ap, bv[tt], acc[tt]);
  }
  // normalize rows, store context as [token][H*D] f16 for the output GEMM
#pragma unroll
  for (int r = 0; r < 8; ++r) {
    const float inv = 1.f / lrow[r];
    const int s = q0 + (l >> 4) * 8 + r;
    _Float16* dst = ctx + (b * 2048 + s) * 640 + h * 80;
#pragma unroll
    for (int tt = 0; tt < 5; ++tt) dst[tt * 16 + (l & 15)] = (_Float16)(acc[tt][r] * inv);
  }
}

// ---- kernel 3: output projection ctx(f16) @ Wo + bo -> f32 ----
__global__ __launch_bounds__(128) void proj_kernel(const _Float16* __restrict__ ctx,
    const _Float16* __restrict__ wot, const float* __restrict__ bo, float* __restrict__ out) {
  const int m0 = blockIdx.x * 64 + threadIdx.y * 16;
  const int n0 = blockIdx.y * 64;
  v8f acc[4] = {};
  gemm_16x64(ctx, m0, wot, n0, acc);
  const int l = threadIdx.x & 31;
#pragma unroll
  for (int j = 0; j < 4; ++j) {
#pragma unroll
    for (int r = 0; r < 8; ++r) {
      const int tok = m0 + (l >> 4) * 8 + r;
      const int c = n0 + j * 16 + (l & 15);
      out[tok * 640 + c] = acc[j][r] + bo[c];
    }
  }
}

extern "C" void kernel_launch(void* const* d_in, const int* in_sizes, int n_in,
                              void* d_out, int out_size, void* d_ws, size_t ws_size,
                              hipStream_t stream) {
  const float* X  = (const float*)d_in[0];   // [2,2048,640]
  const float* Wq = (const float*)d_in[1];
  const float* Wk = (const float*)d_in[2];
  const float* Wv = (const float*)d_in[3];
  const float* Wo = (const float*)d_in[4];
  const float* bo = (const float*)d_in[5];
  float* out = (float*)d_out;

  char* ws = (char*)d_ws;
  const size_t qkBytes  = (size_t)2 * 8 * 2048 * 96 * sizeof(_Float16);   // 6 MB each
  const size_t ctxBytes = (size_t)4096 * 640 * sizeof(_Float16);          // 5.2 MB
  _Float16* q16  = (_Float16*)(ws);
  _Float16* k16  = (_Float16*)(ws + qkBytes);
  _Float16* vt16 = (_Float16*)(ws + 2 * qkBytes);
  _Float16* ctx  = (_Float16*)(ws + 3 * qkBytes);
  _Float16* x16  = (_Float16*)(ws + 3 * qkBytes + ctxBytes);
  _Float16* wt   = (_Float16*)(ws + 3 * qkBytes + 2 * ctxBytes);          // 4 x 640x640 f16

  // zero head-dim pad (d=80..95) so the QK^T K=96 chunks are exact
  hipMemsetAsync(ws, 0, 3 * qkBytes, stream);
  cvt_x_kernel<<<dim3(2560), dim3(256), 0, stream>>>(X, x16);
  wt_kernel<<<dim3(40, 40, 4), dim3(16, 16), 0, stream>>>(Wq, Wk, Wv, Wo, wt);
  qkv_kernel<<<dim3(64, 30), dim3(32, 4), 0, stream>>>(x16, wt, q16, k16, vt16);
  attn_kernel<<<dim3(512), dim3(32, 4), 0, stream>>>(q16, k16, vt16, ctx);
  proj_kernel<<<dim3(64, 10), dim3(32, 4), 0, stream>>>(ctx, wt + 3 * 640 * 640, bo, out);
}
